// TransformerBlock_3573412790757
// MI455X (gfx1250) — matrix operand order repaired and checked by reading, not compiled, not run on hardware
//
#include <hip/hip_runtime.h>
#include <hip/hip_bf16.h>
#include <math.h>

// ---------------------------------------------------------------------------
// Transformer block for MI455X (gfx1250): all GEMMs via v_wmma_f32_16x16x32_bf16
// ---------------------------------------------------------------------------

typedef __attribute__((ext_vector_type(16))) __bf16 v16bf;
typedef __attribute__((ext_vector_type(8)))  float  v8f;

union FragBF { v16bf v; unsigned int u[8]; };

constexpr int B = 8, S = 1024, D = 1024, F = 4096, H = 16, DK = 64;
constexpr int BS = B * S; // 8192 rows

__device__ __forceinline__ unsigned short f2bf(float f) {
  unsigned int u = __float_as_uint(f);
  unsigned int r = (u + 0x7FFFu + ((u >> 16) & 1u)) >> 16;
  return (unsigned short)r;
}

// ---------------- weight transpose + f32->bf16 convert: WT[n][k] = W[k][n] ---
__global__ void k_wtrans(const float* __restrict__ W, unsigned short* __restrict__ WT,
                         int K, int N) {
  __shared__ float t[32][33];
  int n = blockIdx.x * 32 + threadIdx.x;
  int kbase = blockIdx.y * 32;
  for (int i = 0; i < 4; ++i) {
    int k = kbase + threadIdx.y + i * 8;
    t[threadIdx.y + i * 8][threadIdx.x] = W[(size_t)k * N + n];
  }
  __syncthreads();
  for (int i = 0; i < 4; ++i) {
    int n2 = blockIdx.x * 32 + threadIdx.y + i * 8;
    int k2 = kbase + threadIdx.x;
    WT[(size_t)n2 * K + k2] = f2bf(t[threadIdx.x][threadIdx.y + i * 8]);
  }
}

// ---------------- LayerNorm: f32 row -> bf16 row --------------------------
__global__ __launch_bounds__(256)
void k_layernorm(const float* __restrict__ x, const float* __restrict__ g,
                 const float* __restrict__ bta, unsigned short* __restrict__ ybf) {
  __shared__ float s1[256], s2[256];
  int row = blockIdx.x;
  const float* xr = x + (size_t)row * D;
  float v[4];
  float sum = 0.f, sq = 0.f;
  for (int i = 0; i < 4; ++i) {
    v[i] = xr[threadIdx.x + i * 256];
    sum += v[i]; sq += v[i] * v[i];
  }
  s1[threadIdx.x] = sum; s2[threadIdx.x] = sq;
  __syncthreads();
  for (int off = 128; off > 0; off >>= 1) {
    if (threadIdx.x < off) {
      s1[threadIdx.x] += s1[threadIdx.x + off];
      s2[threadIdx.x] += s2[threadIdx.x + off];
    }
    __syncthreads();
  }
  float mu  = s1[0] * (1.0f / D);
  float var = s2[0] * (1.0f / D) - mu * mu;
  float rs  = rsqrtf(var + 1e-5f);
  unsigned short* yr = ybf + (size_t)row * D;
  for (int i = 0; i < 4; ++i) {
    int c = threadIdx.x + i * 256;
    yr[c] = f2bf((v[i] - mu) * rs * g[c] + bta[c]);
  }
}

// ---------------- WMMA GEMM: C[M,N] = A[M,K](bf16) * WT[N,K](bf16) + bias ----
// 128x64 C tile per 256-thread block; each wave owns a 32x32 sub-tile
// (2 A-frags x 2 B-frags -> 4 WMMAs per K-step, 8 LDS dwords per WMMA).
enum { MODE_QKV = 0, MODE_RES = 1, MODE_GELU = 2 };

template <int MODE>
__global__ __launch_bounds__(256)
void k_gemm(const unsigned short* __restrict__ A,
            const unsigned short* __restrict__ WT,
            const float* __restrict__ bias,
            const float* __restrict__ res,      // MODE_RES: f32 residual [M,N]
            float* __restrict__ outF,           // MODE_RES
            unsigned short* __restrict__ outB,  // MODE_QKV / MODE_GELU
            int N, int K) {
  __shared__ unsigned int lA[128 * 16];  // 128 rows x 32 bf16 (8 KB)
  __shared__ unsigned int lB[64 * 16];   //  64 rows x 32 bf16 (4 KB)
  const int tid  = threadIdx.x;
  const int w    = tid >> 5;
  const int lane = tid & 31;
  const int r    = lane & 15, hi = lane >> 4;
  const int rg   = w & 3;            // 32-row group within 128-row tile
  const int cg   = w >> 2;           // 32-col group within 64-col tile
  const int m0   = blockIdx.y * 128, n0 = blockIdx.x * 64;

  v8f c00 = {}, c01 = {}, c10 = {}, c11 = {};
  const int rs4 = K >> 3;  // uint4 per global row
  for (int k0 = 0; k0 < K; k0 += 32) {
    __syncthreads();
    {
      const uint4* ga = (const uint4*)(A  + (size_t)m0 * K + k0);
      const uint4* gb = (const uint4*)(WT + (size_t)n0 * K + k0);
      for (int i = 0; i < 2; ++i) {
        int e = tid + i * 256;            // 512 uint4 for A tile
        int row = e >> 2, c = e & 3;      // 4 uint4 per 32-bf16 row
        ((uint4*)lA)[e] = ga[row * rs4 + c];
      }
      {
        int e = tid;                      // 256 uint4 for B tile
        int row = e >> 2, c = e & 3;
        ((uint4*)lB)[e] = gb[row * rs4 + c];
      }
    }
    __syncthreads();
    FragBF a0, a1, b0, b1;
    for (int j = 0; j < 8; ++j) {
      // A: k = 2*j + 16*hi -> dword j + 8*hi
      int adw = j + 8 * hi;
      a0.u[j] = lA[(rg * 32 + r) * 16 + adw];
      a1.u[j] = lA[(rg * 32 + 16 + r) * 16 + adw];
      // B: k = 2*j + 16*hi -> dword j + 8*hi ; lane r = column
      int bdw = j + 8 * hi;
      b0.u[j] = lB[(cg * 32 + r) * 16 + bdw];
      b1.u[j] = lB[(cg * 32 + 16 + r) * 16 + bdw];
    }
    c00 = __builtin_amdgcn_wmma_f32_16x16x32_bf16(false, a0.v, false, b0.v, (short)0, c00, false, false);
    c01 = __builtin_amdgcn_wmma_f32_16x16x32_bf16(false, a0.v, false, b1.v, (short)0, c01, false, false);
    c10 = __builtin_amdgcn_wmma_f32_16x16x32_bf16(false, a1.v, false, b0.v, (short)0, c10, false, false);
    c11 = __builtin_amdgcn_wmma_f32_16x16x32_bf16(false, a1.v, false, b1.v, (short)0, c11, false, false);
  }

  for (int ti = 0; ti < 2; ++ti)
    for (int tj = 0; tj < 2; ++tj) {
      v8f c = ti ? (tj ? c11 : c10) : (tj ? c01 : c00);
      for (int j = 0; j < 8; ++j) {
        int m = m0 + rg * 32 + ti * 16 + j + 8 * hi;
        int n = n0 + cg * 32 + tj * 16 + r;
        float acc = c[j] + bias[n];
        if (MODE == MODE_QKV) {
          int bb = m / S, s = m % S;
          int h  = n / DK, d = n % DK;
          outB[(((size_t)bb * H + h) * S + s) * DK + d] = f2bf(acc);
        } else if (MODE == MODE_RES) {
          size_t idx = (size_t)m * N + n;
          outF[idx] = res[idx] + acc;
        } else { // MODE_GELU
          float xg = acc;
          float inner = 0.7978845608028654f * (xg + 0.044715f * xg * xg * xg);
          float gl = 0.5f * xg * (1.0f + tanhf(inner));
          outB[(size_t)m * N + n] = f2bf(gl);
        }
      }
    }
}

// ---------------- Flash attention (per (b,h), 64 query rows per block) -------
// q,k,v are bf16 [B,H,S,64]; out abf is bf16 [B,S,D] (heads re-interleaved).
__global__ __launch_bounds__(128)
void k_attention(const unsigned short* __restrict__ qbf,
                 const unsigned short* __restrict__ kbf,
                 const unsigned short* __restrict__ vbf,
                 const int* __restrict__ mask,
                 unsigned short* __restrict__ abf) {
  __shared__ unsigned int lK[32 * 32];     // [key][d/2]  : 32 keys x 64 bf16
  __shared__ unsigned int lVT[64 * 16];    // [d][key/2]  : 64 d   x 32 keys
  __shared__ unsigned int lP[4][16 * 16];  // per-wave P staging 16x32 bf16

  const int tid  = threadIdx.x;
  const int w    = tid >> 5;
  const int lane = tid & 31;
  const int r    = lane & 15, hi = lane >> 4;

  const int ntile = S / 64;
  const int bh   = blockIdx.x / ntile;
  const int tile = blockIdx.x % ntile;
  const int b = bh / H, h = bh % H;
  const int q0 = tile * 64;
  const int qr0 = q0 + w * 16;   // this wave's 16 query rows

  // Q: two A-fragments (dk = 64 = 2 x K32), straight from global
  FragBF aq0, aq1;
  {
    const unsigned int* qp = (const unsigned int*)(qbf + ((size_t)bh * S + qr0) * DK);
    for (int j = 0; j < 8; ++j) {
      int k = 2 * j + hi * 16;
      aq0.u[j] = qp[r * 32 + (k >> 1)];
      aq1.u[j] = qp[r * 32 + ((k + 32) >> 1)];
    }
  }

  float m_run[8], l_run[8];
  v8f O[4] = {};
  for (int j = 0; j < 8; ++j) { m_run[j] = -INFINITY; l_run[j] = 0.f; }

  const unsigned int* kbase = (const unsigned int*)(kbf + (size_t)bh * S * DK);
  const unsigned int* vbase = (const unsigned int*)(vbf + (size_t)bh * S * DK);
  const int* mbase = mask + (size_t)b * S * S;

  for (int kb = 0; kb < S / 32; ++kb) {
    __syncthreads();
    {
      // K tile: 32 keys x 64 bf16, row-major
      const uint4* gk = (const uint4*)(kbase + (size_t)kb * 1024);
      uint4* lk4 = (uint4*)lK;
      for (int i = 0; i < 2; ++i) { int e = tid + i * 128; lk4[e] = gk[e]; }
      // V tile transposed into [d][key]
      unsigned short* vt = (unsigned short*)lVT;
      for (int i = 0; i < 8; ++i) {
        int e = tid + i * 128;       // 1024 dwords
        int key = e >> 5, dp = e & 31;
        unsigned int u = vbase[(size_t)(kb * 32 + key) * 32 + dp];
        vt[(dp * 2) * 32 + key]     = (unsigned short)(u & 0xffffu);
        vt[(dp * 2 + 1) * 32 + key] = (unsigned short)(u >> 16);
      }
    }
    __syncthreads();

    // scores for 32 keys: two 16-key C tiles, each = 2 chained WMMAs over dk
    v8f sc[2];
    for (int kt = 0; kt < 2; ++kt) {
      FragBF bk0, bk1;
      for (int j = 0; j < 8; ++j) {
        bk0.u[j] = lK[(kt * 16 + r) * 32 + j + 8 * hi];        // d 0..31
        bk1.u[j] = lK[(kt * 16 + r) * 32 + j + 8 * hi + 16];   // d 32..63
      }
      v8f c = {};
      c = __builtin_amdgcn_wmma_f32_16x16x32_bf16(false, aq0.v, false, bk0.v, (short)0, c, false, false);
      c = __builtin_amdgcn_wmma_f32_16x16x32_bf16(false, aq1.v, false, bk1.v, (short)0, c, false, false);
      for (int j = 0; j < 8; ++j) {
        int qrow = qr0 + j + 8 * hi;
        int kcol = kb * 32 + kt * 16 + r;
        int mv = mbase[(size_t)qrow * S + kcol];
        float sv = c[j] * 0.125f;                 // 1/sqrt(64)
        c[j] = mv ? sv : -1e9f;
      }
      sc[kt] = c;
    }

    // online softmax (16-lane reductions stay within each wave32 half)
    float p0[8], p1[8], corr[8];
    for (int j = 0; j < 8; ++j) {
      float t = fmaxf(sc[0][j], sc[1][j]);
      for (int off = 1; off < 16; off <<= 1) t = fmaxf(t, __shfl_xor(t, off, 32));
      float newm = fmaxf(m_run[j], t);
      corr[j] = __expf(m_run[j] - newm);
      p0[j] = __expf(sc[0][j] - newm);
      p1[j] = __expf(sc[1][j] - newm);
      float rs = p0[j] + p1[j];
      for (int off = 1; off < 16; off <<= 1) rs += __shfl_xor(rs, off, 32);
      l_run[j] = l_run[j] * corr[j] + rs;
      m_run[j] = newm;
    }
    for (int dt = 0; dt < 4; ++dt)
      for (int j = 0; j < 8; ++j) O[dt][j] *= corr[j];

    // P (C-layout f32) -> bf16 A-fragment via per-wave LDS (LDS in-order per wave)
    {
      unsigned short* ps = (unsigned short*)lP[w];
      for (int j = 0; j < 8; ++j) {
        int row = j + 8 * hi;
        ps[row * 32 + r]      = f2bf(p0[j]);
        ps[row * 32 + 16 + r] = f2bf(p1[j]);
      }
    }
    FragBF pa;
    for (int j = 0; j < 8; ++j)
      pa.u[j] = lP[w][r * 16 + j + 8 * hi];

    for (int dt = 0; dt < 4; ++dt) {
      FragBF bv;
      for (int j = 0; j < 8; ++j)
        bv.u[j] = lVT[(dt * 16 + r) * 16 + j + 8 * hi];
      O[dt] = __builtin_amdgcn_wmma_f32_16x16x32_bf16(false, pa.v, false, bv.v, (short)0, O[dt], false, false);
    }
  }

  // normalize + store back into [B,S,D] with head interleave
  for (int dt = 0; dt < 4; ++dt)
    for (int j = 0; j < 8; ++j) {
      int qrow = qr0 + j + 8 * hi;
      int d = h * DK + dt * 16 + r;
      abf[((size_t)b * S + qrow) * D + d] = f2bf(O[dt][j] / l_run[j]);
    }
}

// ---------------------------------------------------------------------------
extern "C" void kernel_launch(void* const* d_in, const int* in_sizes, int n_in,
                              void* d_out, int out_size, void* d_ws, size_t ws_size,
                              hipStream_t stream) {
  (void)in_sizes; (void)n_in; (void)out_size; (void)ws_size;
  const float* x    = (const float*)d_in[0];
  const int*   mask = (const int*)d_in[1];
  const float* Wq = (const float*)d_in[2];
  const float* bq = (const float*)d_in[3];
  const float* Wk = (const float*)d_in[4];
  const float* bk = (const float*)d_in[5];
  const float* Wv = (const float*)d_in[6];
  const float* bv = (const float*)d_in[7];
  const float* Wo = (const float*)d_in[8];
  const float* bo = (const float*)d_in[9];
  const float* W1 = (const float*)d_in[10];
  const float* b1 = (const float*)d_in[11];
  const float* W2 = (const float*)d_in[12];
  const float* b2 = (const float*)d_in[13];
  const float* ln1g = (const float*)d_in[14];
  const float* ln1b = (const float*)d_in[15];
  const float* ln2g = (const float*)d_in[16];
  const float* ln2b = (const float*)d_in[17];

  // workspace layout (~136 MB, lifetime-overlapped)
  char* ws = (char*)d_ws;
  constexpr size_t SZ_DD = (size_t)D * D * 2;   // 2 MB bf16
  constexpr size_t SZ_DF = (size_t)D * F * 2;   // 8 MB bf16
  unsigned short* WqT = (unsigned short*)(ws);
  unsigned short* WkT = (unsigned short*)(ws + SZ_DD);
  unsigned short* WvT = (unsigned short*)(ws + 2 * SZ_DD);
  unsigned short* WoT = (unsigned short*)(ws + 3 * SZ_DD);
  unsigned short* W1T = (unsigned short*)(ws + 4 * SZ_DD);
  unsigned short* W2T = (unsigned short*)(ws + 4 * SZ_DD + SZ_DF);
  size_t off = 4 * SZ_DD + 2 * SZ_DF;                               // 24 MB
  float* x1 = (float*)(ws + off);              off += (size_t)BS * D * 4;  // 32 MB
  unsigned short* ybf = (unsigned short*)(ws + off); off += (size_t)BS * D * 2; // 16 MB
  unsigned short* qb  = (unsigned short*)(ws + off);       // q,k,v,a: 64 MB total
  unsigned short* kb2 = qb  + (size_t)BS * D;
  unsigned short* vb2 = kb2 + (size_t)BS * D;
  unsigned short* ab  = vb2 + (size_t)BS * D;
  unsigned short* hbf = qb;                    // reuse q..a region (64 MB) for h

  dim3 tb(32, 8);
  k_wtrans<<<dim3(D / 32, D / 32), tb, 0, stream>>>(Wq, WqT, D, D);
  k_wtrans<<<dim3(D / 32, D / 32), tb, 0, stream>>>(Wk, WkT, D, D);
  k_wtrans<<<dim3(D / 32, D / 32), tb, 0, stream>>>(Wv, WvT, D, D);
  k_wtrans<<<dim3(D / 32, D / 32), tb, 0, stream>>>(Wo, WoT, D, D);
  k_wtrans<<<dim3(F / 32, D / 32), tb, 0, stream>>>(W1, W1T, D, F);
  k_wtrans<<<dim3(D / 32, F / 32), tb, 0, stream>>>(W2, W2T, F, D);

  // sublayer 1
  k_layernorm<<<BS, 256, 0, stream>>>(x, ln1g, ln1b, ybf);
  dim3 gD(D / 64, BS / 128);
  k_gemm<MODE_QKV><<<gD, 256, 0, stream>>>(ybf, WqT, bq, nullptr, nullptr, qb,  D, D);
  k_gemm<MODE_QKV><<<gD, 256, 0, stream>>>(ybf, WkT, bk, nullptr, nullptr, kb2, D, D);
  k_gemm<MODE_QKV><<<gD, 256, 0, stream>>>(ybf, WvT, bv, nullptr, nullptr, vb2, D, D);
  k_attention<<<B * H * (S / 64), 128, 0, stream>>>(qb, kb2, vb2, mask, ab);
  k_gemm<MODE_RES><<<gD, 256, 0, stream>>>(ab, WoT, bo, x, x1, nullptr, D, D);

  // sublayer 2
  k_layernorm<<<BS, 256, 0, stream>>>(x1, ln2g, ln2b, ybf);
  k_gemm<MODE_GELU><<<dim3(F / 64, BS / 128), 256, 0, stream>>>(ybf, W1T, b1, nullptr, nullptr, hbf, F, D);
  k_gemm<MODE_RES><<<gD, 256, 0, stream>>>(hbf, W2T, b2, x1, (float*)d_out, nullptr, D, F);
}
